// GraphSAGE_46153718563001
// MI455X (gfx1250) — matrix operand order repaired and checked by reading, not compiled, not run on hardware
//
#include <hip/hip_runtime.h>
#include <hip/hip_bf16.h>

typedef __attribute__((ext_vector_type(16))) __bf16 v16bf;
typedef __attribute__((ext_vector_type(8)))  float  v8f;

#define D 128
#define N0c 1362944
#define N1c 123904
#define N2c 11264
#define N3c 1024
#define E0c 1239040
#define E1c 112640
#define E2c 10240

// ---------------------------------------------------------------------------
// Convert a 128x128 f32 weight (row-major, W[k][n]) into bf16 transposed
// layout WT[n][k] so a WMMA lane can read two runs of 8 consecutive K values of its
// output column (K = half*8 + [0..7], 16 + half*8 + [0..7]) as contiguous 16-byte reads.
// ---------------------------------------------------------------------------
__global__ __launch_bounds__(256) void wconv_kernel(const float* __restrict__ W,
                                                    __bf16* __restrict__ WT) {
    int t = blockIdx.x * 256 + threadIdx.x;   // 16384 threads total
    int n = t & 127;
    int k = t >> 7;
    WT[n * D + k] = (__bf16)W[k * D + n];
}

// ---------------------------------------------------------------------------
// Edge scatter: one wave (32 lanes) per edge. Lane l moves floats
// [4l, 4l+3] of h[src] into msg[dst] via global_atomic_add_f32.
// Lane 0 accumulates the degree.
// ---------------------------------------------------------------------------
__global__ __launch_bounds__(256) void sage_scatter_kernel(
    const float* __restrict__ h, const int* __restrict__ src,
    const int* __restrict__ dst, float* __restrict__ msg,
    float* __restrict__ deg, int nE) {
    int wave = (int)((blockIdx.x * 256 + threadIdx.x) >> 5);
    int lane = threadIdx.x & 31;
    if (wave >= nE) return;
    int s = src[wave];
    int d = dst[wave];
    const float4 v =
        reinterpret_cast<const float4*>(h + (long)s * D)[lane];  // 16B coalesced
    float* p = msg + (long)d * D + lane * 4;
    atomicAdd(p + 0, v.x);
    atomicAdd(p + 1, v.y);
    atomicAdd(p + 2, v.z);
    atomicAdd(p + 3, v.w);
    if (lane == 0) atomicAdd(deg + d, 1.0f);
}

// ---------------------------------------------------------------------------
// Fused SAGE layer GEMM: out = [relu]( (msg/deg) @ Wl + b + h_in @ Wr )
// Block: 256 threads (8 waves) -> 16 rows x 128 cols of output.
// Each wave computes one 16x16 tile with v_wmma_f32_16x16x32_bf16
// (4 K-steps x 2 GEMMs = 8 WMMAs).
// ---------------------------------------------------------------------------
__global__ __launch_bounds__(256) void sage_gemm_kernel(
    const float* __restrict__ msg, const float* __restrict__ deg,
    const float* __restrict__ h_in, const __bf16* __restrict__ WlT,
    const float* __restrict__ bias, const __bf16* __restrict__ WrT,
    float* __restrict__ out, int n_dst, int do_relu) {
    // +8 bf16 row pad (16B) -> lanes 0..15 hit disjoint LDS bank groups
    __shared__ __bf16 smM[16][D + 8];
    __shared__ __bf16 smH[16][D + 8];

    const int t  = threadIdx.x;
    const int m0 = blockIdx.x * 16;

    // ---- Phase 1: stage mean tile (bf16) and h tile (bf16) into LDS ----
    {
        int r   = t >> 4;          // 0..15 row
        int c   = (t & 15) * 8;    // 8 floats per thread
        int row = m0 + r;
        float invd = 0.0f;
        float4 a0 = make_float4(0.f, 0.f, 0.f, 0.f), a1 = a0, b0 = a0, b1 = a0;
        if (row < n_dst) {
            invd = 1.0f / fmaxf(deg[row], 1.0f);
            const float4* pm = reinterpret_cast<const float4*>(msg + (long)row * D + c);
            a0 = pm[0]; a1 = pm[1];
            const float4* ph = reinterpret_cast<const float4*>(h_in + (long)row * D + c);
            b0 = ph[0]; b1 = ph[1];
        }
        __bf16* dm = &smM[r][c];
        dm[0] = (__bf16)(a0.x * invd); dm[1] = (__bf16)(a0.y * invd);
        dm[2] = (__bf16)(a0.z * invd); dm[3] = (__bf16)(a0.w * invd);
        dm[4] = (__bf16)(a1.x * invd); dm[5] = (__bf16)(a1.y * invd);
        dm[6] = (__bf16)(a1.z * invd); dm[7] = (__bf16)(a1.w * invd);
        __bf16* dh = &smH[r][c];
        dh[0] = (__bf16)b0.x; dh[1] = (__bf16)b0.y;
        dh[2] = (__bf16)b0.z; dh[3] = (__bf16)b0.w;
        dh[4] = (__bf16)b1.x; dh[5] = (__bf16)b1.y;
        dh[6] = (__bf16)b1.z; dh[7] = (__bf16)b1.w;
    }
    __syncthreads();

    // ---- Phase 2: WMMA ----
    const int wave = t >> 5;
    const int lane = t & 31;
    const int M    = lane & 15;   // A row / C-D column index
    const int half = lane >> 4;   // lane half selects K sub-range
    const int n0   = wave * 16;
    const int col  = n0 + M;      // output column for this lane

    v8f acc;
    {
        float bv = bias[col];
#pragma unroll
        for (int i = 0; i < 8; ++i) acc[i] = bv;
    }

#pragma unroll
    for (int kb = 0; kb < 4; ++kb) {
        const int kbase = kb * 32;
        v16bf aM, aH, bL, bR;
        // A layout (ISA 16-bit A 16x32): lane=M, VGPR0-3 hold K = half*8 + [0..7],
        // VGPR4-7 hold K = 16 + half*8 + [0..7]
        const __bf16* pm0 = &smM[M][kbase + half * 8];
        const __bf16* pm1 = &smM[M][kbase + 16 + half * 8];
        const __bf16* ph0 = &smH[M][kbase + half * 8];
        const __bf16* ph1 = &smH[M][kbase + 16 + half * 8];
#pragma unroll
        for (int i = 0; i < 8; ++i) {
            aM[i] = pm0[i]; aM[8 + i] = pm1[i];
            aH[i] = ph0[i]; aH[8 + i] = ph1[i];
        }
        // B layout: lane holds column N; elems 0-7 K = half*8 + [0..7], elems 8-15 K = 16 + half*8 + [0..7]
        const __bf16* pl = &WlT[(long)col * D + kbase + half * 8];
        const __bf16* pr = &WrT[(long)col * D + kbase + half * 8];
#pragma unroll
        for (int i = 0; i < 8; ++i) { bL[i] = pl[i]; bR[i] = pr[i]; bL[8 + i] = pl[16 + i]; bR[8 + i] = pr[16 + i]; }

        acc = __builtin_amdgcn_wmma_f32_16x16x32_bf16(false, aM, false, bL,
                                                      (short)0, acc, false, false);
        acc = __builtin_amdgcn_wmma_f32_16x16x32_bf16(false, aH, false, bR,
                                                      (short)0, acc, false, false);
    }

    if (do_relu) {
#pragma unroll
        for (int i = 0; i < 8; ++i) acc[i] = fmaxf(acc[i], 0.0f);
    }

    // C/D layout: VGPR r -> output row m0 + r + half*8, column col
#pragma unroll
    for (int r = 0; r < 8; ++r) {
        int row = m0 + r + half * 8;
        if (row < n_dst) out[(long)row * D + col] = acc[r];
    }
}

// ---------------------------------------------------------------------------
extern "C" void kernel_launch(void* const* d_in, const int* in_sizes, int n_in,
                              void* d_out, int out_size, void* d_ws, size_t ws_size,
                              hipStream_t stream) {
    const float* x    = (const float*)d_in[0];
    const int*   src0 = (const int*)d_in[1];
    const int*   dst0 = (const int*)d_in[2];
    const int*   src1 = (const int*)d_in[3];
    const int*   dst1 = (const int*)d_in[4];
    const int*   src2 = (const int*)d_in[5];
    const int*   dst2 = (const int*)d_in[6];
    const float* Wl0 = (const float*)d_in[7];
    const float* bl0 = (const float*)d_in[8];
    const float* Wr0 = (const float*)d_in[9];
    const float* Wl1 = (const float*)d_in[10];
    const float* bl1 = (const float*)d_in[11];
    const float* Wr1 = (const float*)d_in[12];
    const float* Wl2 = (const float*)d_in[13];
    const float* bl2 = (const float*)d_in[14];
    const float* Wr2 = (const float*)d_in[15];

    // Workspace layout (all sizes 256B multiples)
    char* ws = (char*)d_ws;
    size_t off = 0;
    float* msg = (float*)(ws + off); off += (size_t)N1c * D * 4;   // 63.4 MB (reused L1/L2)
    float* dg  = (float*)(ws + off); off += (size_t)N1c * 4;       // reused
    float* h1  = (float*)(ws + off); off += (size_t)N1c * D * 4;   // 63.4 MB
    float* h2  = (float*)(ws + off); off += (size_t)N2c * D * 4;   // 5.8 MB
    __bf16* wT = (__bf16*)(ws + off);                              // 6 x 32 KB
    __bf16* wTl0 = wT + 0 * D * D;
    __bf16* wTr0 = wT + 1 * D * D;
    __bf16* wTl1 = wT + 2 * D * D;
    __bf16* wTr1 = wT + 3 * D * D;
    __bf16* wTl2 = wT + 4 * D * D;
    __bf16* wTr2 = wT + 5 * D * D;

    // Weight transpose + bf16 conversion (tiny)
    wconv_kernel<<<64, 256, 0, stream>>>(Wl0, wTl0);
    wconv_kernel<<<64, 256, 0, stream>>>(Wr0, wTr0);
    wconv_kernel<<<64, 256, 0, stream>>>(Wl1, wTl1);
    wconv_kernel<<<64, 256, 0, stream>>>(Wr1, wTr1);
    wconv_kernel<<<64, 256, 0, stream>>>(Wl2, wTl2);
    wconv_kernel<<<64, 256, 0, stream>>>(Wr2, wTr2);

    // ---- Layer 0: x (N0) -> h1 (N1), relu ----
    hipMemsetAsync(msg, 0, (size_t)N1c * D * 4, stream);
    hipMemsetAsync(dg,  0, (size_t)N1c * 4, stream);
    sage_scatter_kernel<<<(E0c + 7) / 8, 256, 0, stream>>>(x, src0, dst0, msg, dg, E0c);
    sage_gemm_kernel<<<(N1c + 15) / 16, 256, 0, stream>>>(msg, dg, x, wTl0, bl0, wTr0,
                                                          h1, N1c, 1);

    // ---- Layer 1: h1 (N1) -> h2 (N2), relu ----
    hipMemsetAsync(msg, 0, (size_t)N2c * D * 4, stream);
    hipMemsetAsync(dg,  0, (size_t)N2c * 4, stream);
    sage_scatter_kernel<<<(E1c + 7) / 8, 256, 0, stream>>>(h1, src1, dst1, msg, dg, E1c);
    sage_gemm_kernel<<<(N2c + 15) / 16, 256, 0, stream>>>(msg, dg, h1, wTl1, bl1, wTr1,
                                                          h2, N2c, 1);

    // ---- Layer 2: h2 (N2) -> d_out (N3), no relu ----
    hipMemsetAsync(msg, 0, (size_t)N3c * D * 4, stream);
    hipMemsetAsync(dg,  0, (size_t)N3c * 4, stream);
    sage_scatter_kernel<<<(E2c + 7) / 8, 256, 0, stream>>>(h2, src2, dst2, msg, dg, E2c);
    sage_gemm_kernel<<<(N3c + 15) / 16, 256, 0, stream>>>(msg, dg, h2, wTl2, bl2, wTr2,
                                                          (float*)d_out, N3c, 0);
}
